// MetaLinear_20220706030482
// MI455X (gfx1250) — matrix operand order repaired and checked by reading, not compiled, not run on hardware
//
#include <hip/hip_runtime.h>
#include <hip/hip_bf16.h>

typedef __attribute__((ext_vector_type(16))) __bf16 v16bf;
typedef __attribute__((ext_vector_type(8)))  float  v8f;

#define NTOK 4096
#define IN   256
#define OUT  256
#define EPS  1e-5f

// ---------------- prep: f32 -> bf16 convert (Ww, 16.7M elems) ----------------
__global__ void cvt_bf16_kernel(const float* __restrict__ src,
                                __bf16* __restrict__ dst, int n) {
    int i = blockIdx.x * blockDim.x + threadIdx.x;
    if (i < n) dst[i] = (__bf16)src[i];
}

// ---------------- prep: x -> bf16 and per-token sum(x) ----------------
__global__ void prep_x_kernel(const float* __restrict__ x,
                              __bf16* __restrict__ xbf,
                              float* __restrict__ sx) {
    __shared__ float red[IN];
    const int t = blockIdx.x, i = threadIdx.x;
    float v = x[t * IN + i];
    xbf[t * IN + i] = (__bf16)v;
    red[i] = v;
    __syncthreads();
    for (int s = 128; s > 0; s >>= 1) {
        if (i < s) red[i] += red[i + s];
        __syncthreads();
    }
    if (i == 0) sx[t] = red[0];
}

// ---------------- small path: b = LayerNorm(x @ Wb^T + bb) ----------------
__global__ void bias_ln_kernel(const float* __restrict__ x,
                               const float* __restrict__ Wb,
                               const float* __restrict__ bb,
                               float* __restrict__ bln) {
    __shared__ float xs[IN];
    __shared__ float red[OUT];
    const int t = blockIdx.x, o = threadIdx.x;
    xs[o] = x[t * IN + o];
    __syncthreads();
    float acc = bb[o];
    const float* wrow = Wb + (size_t)o * IN;
    #pragma unroll 4
    for (int j = 0; j < IN; ++j) acc += wrow[j] * xs[j];
    red[o] = acc;
    __syncthreads();
    for (int s = 128; s > 0; s >>= 1) {
        if (o < s) red[o] += red[o + s];
        __syncthreads();
    }
    float mu = red[0] * (1.f / OUT);
    __syncthreads();
    float d = acc - mu;
    red[o] = d * d;
    __syncthreads();
    for (int s = 128; s > 0; s >>= 1) {
        if (o < s) red[o] += red[o + s];
        __syncthreads();
    }
    float var = red[0] * (1.f / OUT);
    bln[t * OUT + o] = d * rsqrtf(var + EPS);
}

// ---------------- main fused WMMA kernel ----------------
// Grid: (NTOK/16, 4). Block: 256 threads = 8 waves.
// Wave computes P-tiles [16 rows x 16 tokens] of P = Ww_bf16 @ Xbf^T (+bw),
// folding LN + x-weighted row reduction without materializing w.
__global__ __launch_bounds__(256, 1)
void meta_wmma_kernel(const float*  __restrict__ x,
                      const __bf16* __restrict__ Wbf,
                      const float*  __restrict__ bw,
                      const __bf16* __restrict__ xbf,
                      const float*  __restrict__ sx,
                      const float*  __restrict__ bln,
                      float* __restrict__ out) {
    __shared__ float xw[IN * 16];            // x transposed: [i][token-in-tile]

    const int tid   = threadIdx.x;
    const int wave  = tid >> 5;
    const int lane  = tid & 31;
    const int tcol  = lane & 15;             // token column / A-row within tile
    const int khalf = lane >> 4;             // which K-half this lane holds
    const int t0    = blockIdx.x * 16;

    // Stage x tile transposed into LDS (f32, used for the x-weighted reduce).
    for (int idx = tid; idx < 16 * IN; idx += 256) {
        const int tk = idx >> 8;             // token 0..15
        const int i  = idx & 255;            // feature
        xw[i * 16 + tk] = x[(size_t)(t0 + tk) * IN + i];
    }
    __syncthreads();

    union BfVec { uint4 u[2]; v16bf v; };

    // Preload B = Xbf token tile: 8 chunks of K=32 (kept in registers, reused
    // for all 128 row-tiles this wave processes).
    // B k order used here (contiguous, same as A): lane n<16 -> col n, K=k0..k0+15; lanes 16-31 -> K=k0+16..k0+31.
    BfVec breg[8];
    const char* xb = (const char*)xbf;
    #pragma unroll
    for (int kc = 0; kc < 8; ++kc) {
        const long off = ((long)(t0 + tcol) * IN + kc * 32 + khalf * 16) * 2;
        breg[kc].u[0] = *(const uint4*)(xb + off);
        breg[kc].u[1] = *(const uint4*)(xb + off + 16);
    }

    const char* wbp   = (const char*)Wbf;
    const int   obase = blockIdx.y * 64 + wave * 8;
    const int   mrow  = khalf * 8;           // first C row (M) held by this lane

    for (int oo = 0; oo < 8; ++oo) {
        const int o = obase + oo;
        float acc0 = 0.f, acc1 = 0.f, acc2 = 0.f;   // sum P, sum P*x, sum P^2

        for (int it = 0; it < 16; ++it) {
            const int rbase = o * IN + it * 16;      // first Ww row of tile
            v8f c = {0.f, 0.f, 0.f, 0.f, 0.f, 0.f, 0.f, 0.f};

            // A k order used here (contiguous, same as B): lane<16 -> row M=lane, K=k0..k0+15;
            // lane>=16 -> same row, K=k0+16..k0+31.
            const long rowoff = (long)(rbase + tcol) * (IN * 2);
            #pragma unroll
            for (int kc = 0; kc < 8; ++kc) {
                BfVec a;
                const char* p = wbp + rowoff + kc * 64 + khalf * 32;
                a.u[0] = *(const uint4*)(p);
                a.u[1] = *(const uint4*)(p + 16);
                c = __builtin_amdgcn_wmma_f32_16x16x32_bf16(
                        false, a.v, false, breg[kc].v, (short)0, c, false, false);
            }

            // bias for this lane's 8 rows (rbase+mrow .. +7)
            const float4* bp = (const float4*)(bw + rbase + mrow);
            const float4 b0 = bp[0], b1 = bp[1];
            const float bias[8] = {b0.x, b0.y, b0.z, b0.w, b1.x, b1.y, b1.z, b1.w};

            const int ibase = it * 16 + mrow;        // feature index of C row 0
            #pragma unroll
            for (int v = 0; v < 8; ++v) {
                const float pv = c[v] + bias[v];
                const float xv = xw[(ibase + v) * 16 + tcol];
                acc0 += pv;
                acc1 += pv * xv;
                acc2 += pv * pv;
            }
        }

        // combine the two lane-halves (rows 0-7 vs 8-15 of each tile)
        acc0 += __shfl_xor(acc0, 16, 32);
        acc1 += __shfl_xor(acc1, 16, 32);
        acc2 += __shfl_xor(acc2, 16, 32);

        const int   t   = t0 + tcol;
        const float mu  = acc0 * (1.f / IN);
        const float var = acc2 * (1.f / IN) - mu * mu;
        const float y   = (acc1 - mu * sx[t]) * rsqrtf(var + EPS)
                          + bln[(size_t)t * OUT + o];
        if (lane < 16) out[(size_t)t * OUT + o] = y;
    }
}

extern "C" void kernel_launch(void* const* d_in, const int* in_sizes, int n_in,
                              void* d_out, int out_size, void* d_ws, size_t ws_size,
                              hipStream_t stream) {
    const float* x  = (const float*)d_in[0];   // [4096,256]
    const float* Ww = (const float*)d_in[1];   // [65536,256]
    const float* bw = (const float*)d_in[2];   // [65536]
    const float* Wb = (const float*)d_in[3];   // [256,256]
    const float* bb = (const float*)d_in[4];   // [256]
    float* out = (float*)d_out;                // [4096,256]

    char* ws = (char*)d_ws;
    __bf16* Wbf = (__bf16*)ws;                                   // 32 MB
    __bf16* xbf = (__bf16*)(ws + 33554432);                      // 2 MB
    float*  sx  = (float*)(ws + 33554432 + 2097152);             // 16 KB
    float*  bln = (float*)(ws + 33554432 + 2097152 + 16384);     // 4 MB

    const int nW = 65536 * IN;
    cvt_bf16_kernel<<<nW / 256, 256, 0, stream>>>(Ww, Wbf, nW);
    prep_x_kernel<<<NTOK, 256, 0, stream>>>(x, xbf, sx);
    bias_ln_kernel<<<NTOK, 256, 0, stream>>>(x, Wb, bb, bln);
    meta_wmma_kernel<<<dim3(NTOK / 16, 4), 256, 0, stream>>>(
        x, Wbf, bw, xbf, sx, bln, out);
}
